// GeneratorBlock_80822694576607
// MI455X (gfx1250) — matrix operand order repaired and checked by reading, not compiled, not run on hardware
//
#include <hip/hip_runtime.h>

typedef __attribute__((ext_vector_type(16))) _Float16 v16h;
typedef __attribute__((ext_vector_type(8)))  _Float16 v8h;
typedef __attribute__((ext_vector_type(8)))  float    v8f;

#define BATCH 16
#define CIN   512
#define COUT  512
#define WDIM  512
#define HH    64
#define WW    64
#define KTAPS 9

// LDS channel-dim padding: 40 halves (80B = 20 dwords) -> conflict-free
// ds_load_b128 across lanes, still 16B-aligned fragment bases.
#define CPAD  40

// ---------------------------------------------------------------------------
// Kernel 1: style[b][i] = w_embs[b] . style_w[i] + style_b[i] + 1
// ---------------------------------------------------------------------------
__global__ void style_kernel(const float* __restrict__ w_embs,
                             const float* __restrict__ style_w,
                             const float* __restrict__ style_b,
                             float* __restrict__ style) {
    int gid = blockIdx.x * blockDim.x + threadIdx.x;
    if (gid >= BATCH * CIN) return;
    int b = gid / CIN, i = gid % CIN;
    const float* we = w_embs + (size_t)b * WDIM;
    const float* sw = style_w + (size_t)i * WDIM;
    float acc = 0.f;
    for (int j = 0; j < WDIM; ++j) acc += we[j] * sw[j];
    style[gid] = acc + style_b[i] + 1.0f;
}

// ---------------------------------------------------------------------------
// Kernel 2: per (b,out) demodulated f16 weights, tap-major layout:
//   wmod[((b*9 + t)*COUT + o)*CIN + ci]
// One 256-thread block per (b,out).
// ---------------------------------------------------------------------------
__global__ void modw_kernel(const float* __restrict__ conv_w,
                            const float* __restrict__ style,
                            _Float16* __restrict__ wmod) {
    const int b = blockIdx.x / COUT;
    const int o = blockIdx.x % COUT;
    const float* wrow = conv_w + (size_t)o * (CIN * KTAPS);
    const float* st   = style + (size_t)b * CIN;

    __shared__ float red[256];
    float ss = 0.f;
    for (int e = threadIdx.x; e < CIN * KTAPS; e += 256) {
        float v = wrow[e] * st[e / 9];
        ss += v * v;
    }
    red[threadIdx.x] = ss;
    __syncthreads();
    for (int s = 128; s > 0; s >>= 1) {
        if ((int)threadIdx.x < s) red[threadIdx.x] += red[threadIdx.x + s];
        __syncthreads();
    }
    const float inv = rsqrtf(red[0]);

    for (int e = threadIdx.x; e < CIN * KTAPS; e += 256) {
        int ci = e / 9, t = e % 9;
        float v = wrow[e] * st[ci] * inv;
        wmod[((size_t)(b * KTAPS + t) * COUT + o) * CIN + ci] = (_Float16)v;
    }
}

// ---------------------------------------------------------------------------
// Kernel 3: implicit-GEMM conv via WMMA f32<-f16 16x16x32, tap-decomposed.
// grid = (HH, COUT/128, BATCH); block = 256 (8 wave32).
// Workgroup tile: 128 out-channels (M, 8 waves x 16) x 64 pixels (N = row y).
// Each wave: 16(M) x 64(N) => 4 accumulators; 1 A-load feeds 4 WMMAs.
// ---------------------------------------------------------------------------
__global__ __launch_bounds__(256)
void conv_kernel(const float* __restrict__ imgs,
                 const _Float16* __restrict__ wmod,
                 const float* __restrict__ conv_b,
                 const float* __restrict__ noise,
                 const float* __restrict__ nwp,
                 float* __restrict__ out) {
    const int y    = blockIdx.x;          // image row
    const int m0   = blockIdx.y * 128;    // out-channel tile base
    const int b    = blockIdx.z;
    const int tid  = threadIdx.x;
    const int lane = tid & 31;
    const int wv   = tid >> 5;            // 0..7
    const int wM   = wv * 16;             // wave M offset in tile

    // LDS staging: xs[row 0..2][col 0..65][chan 0..31 (+pad)], col = x+1 (halo)
    __shared__ _Float16 xs[3 * 66 * CPAD];

    v8f acc[4] = {};                      // N sub-bases 0,16,32,48

    const int mrow  = m0 + wM + (lane & 15);      // A-matrix M row
    const int klo   = (lane < 16) ? 0 : 8;        // A frag K base (ISA layout)
    const int khalf = (lane < 16) ? 0 : 16;       // unused; B frag uses klo (same order as A)
    const int nlane = lane & 15;                  // B/C/D N = lane%16

    for (int c0 = 0; c0 < CIN; c0 += 32) {
        // ---- fill LDS (f32 -> f16 convert), zero-pad halo ----
        for (int idx = tid; idx < 3 * 66 * 32; idx += 256) {
            int col = idx % 66;                   // x+1, coalesced in x
            int rc  = idx / 66;
            int r   = rc % 3;                     // dy+1
            int c   = rc / 3;                     // chunk-local channel
            int x   = col - 1;
            int yy  = y + r - 1;
            float v = 0.f;
            if ((unsigned)x < 64u && (unsigned)yy < 64u)
                v = imgs[(((size_t)b * CIN + c0 + c) * HH + yy) * WW + x];
            xs[(r * 66 + col) * CPAD + c] = (_Float16)v;
        }
        __syncthreads();

        // ---- 9 taps x 4 WMMAs per wave, one A-load per tap ----
        #pragma unroll
        for (int t = 0; t < 9; ++t) {
            const int r  = t / 3;                 // dy+1
            const int dx = t % 3;                 // dx+1

            // A fragment: 16x32 f16 (two b128 runs per lane, ISA layout)
            const _Float16* arow =
                wmod + ((size_t)(b * KTAPS + t) * COUT + mrow) * CIN + c0;
            v8h alo = *(const v8h*)(arow + klo);
            v8h ahi = *(const v8h*)(arow + klo + 16);
            v16h a;
            #pragma unroll
            for (int i = 0; i < 8; ++i) { a[i] = alo[i]; a[8 + i] = ahi[i]; }

            #pragma unroll
            for (int ns = 0; ns < 4; ++ns) {
                int col = ns * 16 + nlane + dx;
                const _Float16* bp = &xs[(r * 66 + col) * CPAD + klo];
                v8h b0 = *(const v8h*)bp;
                v8h b1 = *(const v8h*)(bp + 16);
                v16h bf;
                #pragma unroll
                for (int i = 0; i < 8; ++i) { bf[i] = b0[i]; bf[8 + i] = b1[i]; }
                acc[ns] = __builtin_amdgcn_wmma_f32_16x16x32_f16(
                    false, a, false, bf, (short)0, acc[ns], false, false);
            }
        }
        __syncthreads();
    }

    // ---- epilogue: bias + noise + LeakyReLU; C/D: M = rr + 8*(lane>=16) ----
    const float nw    = nwp[0];
    const int   mhalf = (lane < 16) ? 0 : 8;
    #pragma unroll
    for (int rr = 0; rr < 8; ++rr) {
        const int   m    = m0 + wM + mhalf + rr;
        const float bias = conv_b[m];
        #pragma unroll
        for (int ns = 0; ns < 4; ++ns) {
            int x = ns * 16 + nlane;
            float v = acc[ns][rr] + bias
                    + noise[((size_t)b * HH + y) * WW + x] * nw;
            out[(((size_t)b * COUT + m) * HH + y) * WW + x] =
                v >= 0.f ? v : 0.2f * v;
        }
    }
}

// ---------------------------------------------------------------------------
extern "C" void kernel_launch(void* const* d_in, const int* in_sizes, int n_in,
                              void* d_out, int out_size, void* d_ws, size_t ws_size,
                              hipStream_t stream) {
    (void)in_sizes; (void)n_in; (void)out_size; (void)ws_size;
    const float* imgs    = (const float*)d_in[0];
    const float* w_embs  = (const float*)d_in[1];
    const float* noise   = (const float*)d_in[2];
    const float* conv_w  = (const float*)d_in[3];
    const float* conv_b  = (const float*)d_in[4];
    const float* style_w = (const float*)d_in[5];
    const float* style_b = (const float*)d_in[6];
    const float* nw      = (const float*)d_in[7];
    float* out = (float*)d_out;

    // workspace layout: style f32[16*512] at 0; wmod f16[16*9*512*512] at 64KB
    float*    style = (float*)d_ws;
    _Float16* wmod  = (_Float16*)((char*)d_ws + 65536);

    style_kernel<<<(BATCH * CIN + 255) / 256, 256, 0, stream>>>(
        w_embs, style_w, style_b, style);
    modw_kernel<<<BATCH * COUT, 256, 0, stream>>>(conv_w, style, wmod);
    conv_kernel<<<dim3(HH, COUT / 128, BATCH), 256, 0, stream>>>(
        imgs, wmod, conv_b, noise, nw, out);
}
